// Attention_45664092291505
// MI455X (gfx1250) — matrix operand order repaired and checked by reading, not compiled, not run on hardware
//
#include <hip/hip_runtime.h>

#define BDIM 4
#define SDIM 2048
#define EDIM 256
#define HDIM 8
#define CDIM 32
#define NEG_INF_V 32768.0f
#define LN_EPS 1e-5f
#define L2E 1.4426950408889634f

typedef __attribute__((ext_vector_type(16))) __bf16 bf16x16;
typedef __attribute__((ext_vector_type(8)))  __bf16 bf16x8;
typedef __attribute__((ext_vector_type(8)))  float  f32x8;

static __device__ __forceinline__ f32x8 wmma_bf16(bf16x16 a, bf16x16 b, f32x8 c) {
  // D = A(16x32 bf16) * B(32x16 bf16) + C(16x16 f32)
  return __builtin_amdgcn_wmma_f32_16x16x32_bf16(false, a, false, b, (short)0, c, false, false);
}

// ---------------------------------------------------------------------------
// Kernel 0: one-time weight transpose + fp32->bf16 convert.
// Wt[(m*256 + n)*256 + k] = W_m[k*256 + n],  m in {q,k,v,g,o}.
// Makes every GEMM B-fragment a single contiguous 32B v16bf load.
// ---------------------------------------------------------------------------
__global__ __launch_bounds__(256) void k_wtrans(
    const float* __restrict__ wq, const float* __restrict__ wk,
    const float* __restrict__ wv, const float* __restrict__ wg,
    const float* __restrict__ wo, __bf16* __restrict__ Wt)
{
  const int mk = blockIdx.x;          // 0..1279 = m*256 + k
  const int m = mk >> 8, k = mk & 255;
  const float* W = (m == 0) ? wq : (m == 1) ? wk : (m == 2) ? wv
                   : (m == 3) ? wg : wo;
  const int n = threadIdx.x;          // coalesced read of row k
  Wt[((size_t)(m * 256 + n)) * 256 + k] = (__bf16)W[(size_t)k * 256 + n];
}

// ---------------------------------------------------------------------------
// Kernel 1: LayerNorm + fused QKVG projection (xn @ Wt[0..3]).
// Block: 256 threads (8 waves), 16 tokens. Wave w covers cols [w*128,w*128+128)
// of the 1024-wide concatenated output -> each wave lies in ONE matrix.
// Q,K,G stored [B,H,S,C] bf16; V stored transposed [B,H,C,S] bf16.
// ---------------------------------------------------------------------------
__global__ __launch_bounds__(256) void k_ln_qkvg(
    const float* __restrict__ x, const __bf16* __restrict__ Wt,
    const float* __restrict__ bg,
    __bf16* __restrict__ Q, __bf16* __restrict__ K,
    __bf16* __restrict__ Vt, __bf16* __restrict__ G)
{
  __shared__ float  xs[16 * 260];
  __shared__ __bf16 xbf[16 * 258];   // normalized bf16 tile, A-frag friendly

  const int tid  = threadIdx.x;
  const int lane = tid & 31;
  const int wave = tid >> 5;
  const int row0 = blockIdx.x * 16;

  #pragma unroll
  for (int i = 0; i < 16; ++i) {
    int e = tid + i * 256;
    int r = e >> 8, c = e & 255;
    xs[r * 260 + c] = x[(size_t)(row0 + r) * EDIM + c];
  }
  __syncthreads();

  {
    int r = tid >> 4, s = tid & 15;
    float sum = 0.f, sq = 0.f;
    #pragma unroll
    for (int i = 0; i < 16; ++i) {
      float v = xs[r * 260 + s + 16 * i];
      sum += v; sq += v * v;
    }
    #pragma unroll
    for (int m = 1; m <= 8; m <<= 1) {
      sum += __shfl_xor(sum, m, 32);
      sq  += __shfl_xor(sq,  m, 32);
    }
    float mu  = sum * (1.0f / EDIM);
    float var = sq * (1.0f / EDIM) - mu * mu;
    float rs  = rsqrtf(var + LN_EPS);
    #pragma unroll
    for (int i = 0; i < 16; ++i) {
      int c = s + 16 * i;
      xbf[r * 258 + c] = (__bf16)((xs[r * 260 + c] - mu) * rs);
    }
  }
  __syncthreads();

  const int colbase = wave * 128;
  const int m_  = lane & 15;
  const int grp = lane >> 4;

  f32x8 acc[8];
  #pragma unroll
  for (int t = 0; t < 8; ++t) acc[t] = (f32x8)0.f;

  const unsigned* arow = (const unsigned*)&xbf[m_ * 258];
  for (int k0 = 0; k0 < EDIM; k0 += 32) {
    union { bf16x16 v; unsigned u[8]; } a;
    const int dbase = k0 >> 1;
    const int kb = grp * 8;
    #pragma unroll
    for (int j = 0; j < 4; ++j) {
      a.u[j]     = arow[dbase + kb + j];
      a.u[4 + j] = arow[dbase + kb + 4 + j];
    }
    const int kcol = k0 + grp * 16;   // contiguous K range of this lane's B-frag
    #pragma unroll
    for (int t = 0; t < 8; ++t) {
      bf16x16 bfr = *(const bf16x16*)(Wt + (size_t)(colbase + t * 16 + m_) * 256 + kcol);
      acc[t] = wmma_bf16(a.v, bfr, acc[t]);
    }
  }

  // --- epilogue: mat is wave-uniform (wave>>1); force scalar branch ---
  const int mat = __builtin_amdgcn_readfirstlane(wave >> 1);
  #pragma unroll
  for (int t = 0; t < 8; ++t) {
    const int c255 = (colbase + t * 16 + m_) & 255;
    const int h = c255 >> 5, c = c255 & 31;
    #pragma unroll
    for (int r = 0; r < 8; ++r) {
      const int tok = row0 + grp * 8 + r;
      const int b = tok >> 11, s = tok & (SDIM - 1);
      const size_t bh = (size_t)(b * HDIM + h);
      const float v = acc[t][r];
      if (mat == 0)      Q [(bh * SDIM + s) * CDIM + c] = (__bf16)v;
      else if (mat == 1) K [(bh * SDIM + s) * CDIM + c] = (__bf16)v;
      else if (mat == 2) Vt[(bh * CDIM + c) * SDIM + s] = (__bf16)v;
      else {
        float g = 1.0f / (1.0f + __expf(-(v + bg[c255])));
        G[(bh * SDIM + s) * CDIM + c] = (__bf16)g;
      }
    }
  }
}

// ---------------------------------------------------------------------------
// Kernel 2: fused flash attention, 64-key tiles. One wave = 16 queries of one
// (b,h). 4 logit WMMAs + NT bias stream + online softmax (exp2-based) +
// WMMA-with-ones row-sum + LDS prob transpose + 4 P*V WMMAs per tile.
// ---------------------------------------------------------------------------
__global__ __launch_bounds__(256) void k_attn(
    const __bf16* __restrict__ Q, const __bf16* __restrict__ K,
    const __bf16* __restrict__ Vt, const __bf16* __restrict__ G,
    const float* __restrict__ bias, const int* __restrict__ mask,
    __bf16* __restrict__ Og)
{
  __shared__ __bf16 plds[8][16 * 66];   // per-wave 16x64 prob tile (pad 2)

  const int tid = threadIdx.x, lane = tid & 31, wave = tid >> 5;
  const int bh = blockIdx.y;
  const int b = bh >> 3, h = bh & 7;
  const int i0 = (blockIdx.x * 8 + wave) * 16;
  const int m_ = lane & 15, grp = lane >> 4;
  const int rgrp = grp * 8;
  const int kb2 = grp * 16;

  union { bf16x16 v; bf16x8 hlf[2]; } qa;
  {
    const __bf16* qrow = Q + ((size_t)bh * SDIM + i0 + m_) * CDIM;
    qa.hlf[0] = *(const bf16x8*)(qrow + grp * 16);
    qa.hlf[1] = *(const bf16x8*)(qrow + grp * 16 + 8);
  }

  bf16x16 ones;
  #pragma unroll
  for (int e = 0; e < 16; ++e) ones[e] = (__bf16)1.0f;

  f32x8 o0 = (f32x8)0.f, o1 = (f32x8)0.f;
  float lsum[8], mrow[8];
  #pragma unroll
  for (int r = 0; r < 8; ++r) { lsum[r] = 0.f; mrow[r] = -3.0e38f; }

  const float scl2  = 0.17677669529663687f * L2E;  // (1/sqrt(C)) * log2(e)
  const float maskv = -NEG_INF_V * L2E;
  __bf16* pw = plds[wave];
  const float* br = bias + ((size_t)bh * SDIM + i0 + rgrp) * SDIM;
  const __bf16* vbase = Vt + (size_t)bh * CDIM * SDIM;
  const __bf16* kbase = K + (size_t)bh * SDIM * CDIM;

  for (int jt = 0; jt < SDIM; jt += 64) {
    // ---- logits: 4 WMMAs (64 keys) ----
    bf16x16 kf0 = *(const bf16x16*)(kbase + (size_t)(jt + m_) * CDIM + kb2);
    bf16x16 kf1 = *(const bf16x16*)(kbase + (size_t)(jt + 16 + m_) * CDIM + kb2);
    bf16x16 kf2 = *(const bf16x16*)(kbase + (size_t)(jt + 32 + m_) * CDIM + kb2);
    bf16x16 kf3 = *(const bf16x16*)(kbase + (size_t)(jt + 48 + m_) * CDIM + kb2);
    f32x8 s0 = wmma_bf16(qa.v, kf0, (f32x8)0.f);
    f32x8 s1 = wmma_bf16(qa.v, kf1, (f32x8)0.f);
    f32x8 s2 = wmma_bf16(qa.v, kf2, (f32x8)0.f);
    f32x8 s3 = wmma_bf16(qa.v, kf3, (f32x8)0.f);

    const int j0 = jt + m_;
    const bool keep0 = mask[b * SDIM + j0]      != 0;
    const bool keep1 = mask[b * SDIM + j0 + 16] != 0;
    const bool keep2 = mask[b * SDIM + j0 + 32] != 0;
    const bool keep3 = mask[b * SDIM + j0 + 48] != 0;
    if (jt + 64 < SDIM) __builtin_prefetch(br + j0 + 64, 0, 0);

    float nm[8];
    #pragma unroll
    for (int r = 0; r < 8; ++r) {
      const float* brr = br + (size_t)r * SDIM + j0;
      float b0 = __builtin_nontemporal_load(brr);
      float b1 = __builtin_nontemporal_load(brr + 16);
      float b2 = __builtin_nontemporal_load(brr + 32);
      float b3 = __builtin_nontemporal_load(brr + 48);
      float v0 = keep0 ? fmaf(s0[r], scl2, b0 * L2E) : maskv;
      float v1 = keep1 ? fmaf(s1[r], scl2, b1 * L2E) : maskv;
      float v2 = keep2 ? fmaf(s2[r], scl2, b2 * L2E) : maskv;
      float v3 = keep3 ? fmaf(s3[r], scl2, b3 * L2E) : maskv;
      s0[r] = v0; s1[r] = v1; s2[r] = v2; s3[r] = v3;
      float t = fmaxf(fmaxf(v0, v1), fmaxf(v2, v3));
      t = fmaxf(t, __shfl_xor(t, 1, 32));
      t = fmaxf(t, __shfl_xor(t, 2, 32));
      t = fmaxf(t, __shfl_xor(t, 4, 32));
      t = fmaxf(t, __shfl_xor(t, 8, 32));
      nm[r] = fmaxf(mrow[r], t);
    }
    #pragma unroll
    for (int r = 0; r < 8; ++r) {
      float alpha = __builtin_amdgcn_exp2f(mrow[r] - nm[r]);
      mrow[r] = nm[r];
      float e0 = __builtin_amdgcn_exp2f(s0[r] - nm[r]);
      float e1 = __builtin_amdgcn_exp2f(s1[r] - nm[r]);
      float e2 = __builtin_amdgcn_exp2f(s2[r] - nm[r]);
      float e3 = __builtin_amdgcn_exp2f(s3[r] - nm[r]);
      lsum[r] *= alpha;
      o0[r] *= alpha; o1[r] *= alpha;
      __bf16* pr = pw + (rgrp + r) * 66 + m_;
      pr[0]  = (__bf16)e0;
      pr[16] = (__bf16)e1;
      pr[32] = (__bf16)e2;
      pr[48] = (__bf16)e3;
    }
    // ---- reload probs as two 16x32 A fragments (same-wave DS ordering) ----
    union { bf16x16 v; unsigned u[8]; } pa0, pa1;
    {
      const unsigned* prow = (const unsigned*)(pw + m_ * 66);
      const int kb4 = grp * 8;
      #pragma unroll
      for (int j = 0; j < 4; ++j) {
        pa0.u[j]     = prow[kb4 + j];
        pa0.u[4 + j] = prow[kb4 + 4 + j];
        pa1.u[j]     = prow[16 + kb4 + j];
        pa1.u[4 + j] = prow[20 + kb4 + j];
      }
    }
    // ---- row-sum of probs via WMMA against all-ones B ----
    f32x8 ls = wmma_bf16(pa0.v, ones, (f32x8)0.f);
    ls = wmma_bf16(pa1.v, ones, ls);
    #pragma unroll
    for (int r = 0; r < 8; ++r) lsum[r] += ls[r];

    // ---- O += P * V ----
    bf16x16 vf0a = *(const bf16x16*)(vbase + (size_t)m_ * SDIM + jt + kb2);
    bf16x16 vf1a = *(const bf16x16*)(vbase + (size_t)(16 + m_) * SDIM + jt + kb2);
    o0 = wmma_bf16(pa0.v, vf0a, o0);
    o1 = wmma_bf16(pa0.v, vf1a, o1);
    bf16x16 vf0b = *(const bf16x16*)(vbase + (size_t)m_ * SDIM + jt + 32 + kb2);
    bf16x16 vf1b = *(const bf16x16*)(vbase + (size_t)(16 + m_) * SDIM + jt + 32 + kb2);
    o0 = wmma_bf16(pa1.v, vf0b, o0);
    o1 = wmma_bf16(pa1.v, vf1b, o1);
  }

  // ---- epilogue: out = gate * O / l, stored bf16 as [B,S,H*C] ----
  #pragma unroll
  for (int r = 0; r < 8; ++r) {
    const int i = i0 + rgrp + r;
    const float inv = 1.0f / lsum[r];
    const size_t grow = ((size_t)bh * SDIM + i) * CDIM;
    const float g0 = (float)G[grow + m_];
    const float g1 = (float)G[grow + 16 + m_];
    const size_t orow = (size_t)(b * SDIM + i) * (HDIM * CDIM) + h * CDIM;
    Og[orow + m_]      = (__bf16)(o0[r] * inv * g0);
    Og[orow + 16 + m_] = (__bf16)(o1[r] * inv * g1);
  }
}

// ---------------------------------------------------------------------------
// Kernel 3: output projection out = Og @ wo + bo (bf16 WMMA, fp32 out).
// ---------------------------------------------------------------------------
__global__ __launch_bounds__(256) void k_outproj(
    const __bf16* __restrict__ Og, const __bf16* __restrict__ Wt,
    const float* __restrict__ bo, float* __restrict__ out)
{
  const int tid = threadIdx.x, lane = tid & 31, wave = tid >> 5;
  const int row0 = blockIdx.x * 16;
  const int m_ = lane & 15, grp = lane >> 4;
  const int colbase = wave * 32;
  const int kb = grp * 8, kb2 = grp * 16;
  const __bf16* Wo = Wt + (size_t)4 * 256 * 256;   // transposed wo, bf16

  f32x8 acc0 = (f32x8)0.f, acc1 = (f32x8)0.f;
  const __bf16* arow = Og + (size_t)(row0 + m_) * 256;

  for (int k0 = 0; k0 < 256; k0 += 32) {
    union { bf16x16 v; bf16x8 hlf[2]; } a;
    a.hlf[0] = *(const bf16x8*)(arow + k0 + kb2);
    a.hlf[1] = *(const bf16x8*)(arow + k0 + 8 + kb2);
    bf16x16 b0 = *(const bf16x16*)(Wo + (size_t)(colbase + m_) * 256 + k0 + kb2);
    bf16x16 b1 = *(const bf16x16*)(Wo + (size_t)(colbase + 16 + m_) * 256 + k0 + kb2);
    acc0 = wmma_bf16(a.v, b0, acc0);
    acc1 = wmma_bf16(a.v, b1, acc1);
  }
  const float bo0 = bo[colbase + m_], bo1 = bo[colbase + 16 + m_];
  #pragma unroll
  for (int r = 0; r < 8; ++r) {
    const size_t orow = (size_t)(row0 + grp * 8 + r) * 256;
    __builtin_nontemporal_store(acc0[r] + bo0, out + orow + colbase + m_);
    __builtin_nontemporal_store(acc1[r] + bo1, out + orow + colbase + 16 + m_);
  }
}

// ---------------------------------------------------------------------------
extern "C" void kernel_launch(void* const* d_in, const int* in_sizes, int n_in,
                              void* d_out, int out_size, void* d_ws, size_t ws_size,
                              hipStream_t stream) {
  const float* x    = (const float*)d_in[0];
  const int*   mask = (const int*)d_in[1];
  const float* bias = (const float*)d_in[2];
  const float* wq   = (const float*)d_in[3];
  const float* wk   = (const float*)d_in[4];
  const float* wv   = (const float*)d_in[5];
  const float* wg   = (const float*)d_in[6];
  const float* bg   = (const float*)d_in[7];
  const float* wo   = (const float*)d_in[8];
  const float* bo   = (const float*)d_in[9];
  float* out = (float*)d_out;

  const size_t QELEMS = (size_t)BDIM * HDIM * SDIM * CDIM;   // 2,097,152
  char* ws = (char*)d_ws;
  __bf16* Q  = (__bf16*)(ws);                  // [B,H,S,C]   4 MB
  __bf16* K  = (__bf16*)(ws + QELEMS * 2);     // [B,H,S,C]   4 MB
  __bf16* Vt = (__bf16*)(ws + QELEMS * 4);     // [B,H,C,S]   4 MB
  __bf16* G  = (__bf16*)(ws + QELEMS * 6);     // [B,H,S,C]   4 MB
  __bf16* Og = (__bf16*)(ws + QELEMS * 8);     // [B,S,H*C]   4 MB
  __bf16* Wt = (__bf16*)(ws + QELEMS * 10);    // [5*256,256] 640 KB

  k_wtrans <<<dim3(1280), dim3(256), 0, stream>>>(wq, wk, wv, wg, wo, Wt);
  k_ln_qkvg<<<dim3(512),  dim3(256), 0, stream>>>(x, Wt, bg, Q, K, Vt, G);
  k_attn   <<<dim3(SDIM / 128, BDIM * HDIM), dim3(256), 0, stream>>>(Q, K, Vt, G, bias, mask, Og);
  k_outproj<<<dim3(512),  dim3(256), 0, stream>>>(Og, Wt, bo, out);
}
